// AtariMambaDQNNetwork_77936476553718
// MI455X (gfx1250) — matrix operand order repaired and checked by reading, not compiled, not run on hardware
//
#include <hip/hip_runtime.h>
#include <hip/hip_bf16.h>

// ---------------------------------------------------------------------------
// AtariMambaDQN on MI455X (gfx1250).
//  - All GEMM-shaped work through v_wmma_f32_16x16x32_f16 (wave32).
//  - Convs are implicit GEMM; A fragments use vectorized patch loads.
//  - conv1 weights staged to LDS via global_load_async_to_lds_b128 (ASYNCcnt);
//    conv2/conv3 weights staged via the Tensor Data Mover (tensor_load_to_lds,
//    TENSORcnt) -- one 2D-tile descriptor per workgroup, issued by wave 0.
//  - All B fragments then come from LDS (ds_load).
//  - Sequential selective scan in VALU registers.
// ---------------------------------------------------------------------------

typedef __attribute__((ext_vector_type(16))) _Float16 v16h;
typedef __attribute__((ext_vector_type(8)))  _Float16 v8h;
typedef __attribute__((ext_vector_type(8)))  float    v8f;
typedef __attribute__((ext_vector_type(4)))  unsigned u32x4;
typedef __attribute__((ext_vector_type(8)))  unsigned u32x8;

#define NFRAMES 2048   // B*S = 64*32
#define D_MODEL 128
#define D_INNER 256
#define D_STATE 32
#define DT_RANK 8

__device__ __forceinline__ v8f wmma16(v16h a, v16h b, v8f c) {
  return __builtin_amdgcn_wmma_f32_16x16x32_f16(
      /*neg_a=*/false, a, /*neg_b=*/false, b,
      /*c_mod=*/(short)0, c, /*reuse_a=*/false, /*reuse_b=*/false);
}

__device__ __forceinline__ float sigmoidf_(float x) {
  return 1.0f / (1.0f + __expf(-x));
}

// ---------------------------------------------------------------------------
// Async staging (per-lane b128 copies, ASYNCcnt) of an f16 matrix into LDS.
// ---------------------------------------------------------------------------
__device__ __forceinline__ void stage_weights_lds(_Float16* __restrict__ sh,
                                                  const _Float16* __restrict__ g,
                                                  int nhalf) {
  int nseg = nhalf >> 3;   // 16-byte segments
  for (int i = threadIdx.x; i < nseg; i += blockDim.x) {
    unsigned lds_off = (unsigned)(i << 4);
    unsigned long long ga = (unsigned long long)(const void*)(g + (i << 3));
    asm volatile("global_load_async_to_lds_b128 %0, %1, off"
                 :: "v"(lds_off), "v"(ga) : "memory");
  }
  asm volatile("s_wait_asynccnt 0x0" ::: "memory");
}

// ---------------------------------------------------------------------------
// TDM staging: one 2D-tile descriptor moves the whole (dim1 x dim0) f16
// matrix (row stride = stride0 elements) from global into LDS offset 0.
// Issued by the calling wave only; caller gates to one wave + barriers.
// D# layout per CDNA5 ISA ch.8 (group0 128b, group1 256b; groups 2/3 NULL).
// ---------------------------------------------------------------------------
__device__ __forceinline__ void tdm_stage_weights(const void* gaddr,
                                                  unsigned dim0, unsigned dim1,
                                                  unsigned stride0) {
  unsigned long long ga = (unsigned long long)gaddr;
  u32x4 g0;
  g0[0] = 1u;                                               // count=1, user D#
  g0[1] = 0u;                                               // lds_addr = 0
  g0[2] = (unsigned)ga;                                     // global_addr[31:0]
  g0[3] = (unsigned)((ga >> 32) & 0x1FFFFFFu) | (2u << 30); // addr[56:32], type=2
  u32x8 g1;
  g1[0] = 1u << 16;                          // data_size = 2B; no mask/pad/iter
  g1[1] = (dim0 & 0xFFFFu) << 16;            // tensor_dim0[15:0]
  g1[2] = (dim0 >> 16) | ((dim1 & 0xFFFFu) << 16);   // dim0[31:16] | dim1[15:0]
  g1[3] = (dim1 >> 16) | ((dim0 & 0xFFFFu) << 16);   // dim1[31:16] | tile_dim0
  g1[4] = dim1 & 0xFFFFu;                    // tile_dim1 (tile_dim2 = 0)
  g1[5] = stride0;                           // tensor_dim0_stride[31:0]
  g1[6] = 0u;                                // stride0[47:32] | stride1[15:0]
  g1[7] = 0u;                                // stride1[47:16]
  asm volatile("tensor_load_to_lds %0, %1" :: "s"(g0), "s"(g1) : "memory");
  __builtin_amdgcn_s_wait_tensorcnt(0);
}

// ---------------------------------------------------------------------------
// B fragment from LDS-resident W (N,K) row-major: lane l -> col n0+(l&15),
// halves e -> K = k0 + e + (e&8) + 8*(l>>4). Two 16B ds reads.
// ---------------------------------------------------------------------------
__device__ __forceinline__ v16h lds_b_frag(const _Float16* __restrict__ shW,
                                           int K, int n0, int k0, int lane) {
  int hi = lane >> 4;
  int n  = n0 + (lane & 15);
  v8h b0 = *(const v8h*)(shW + (size_t)n * K + k0 + 8 * hi);
  v8h b1 = *(const v8h*)(shW + (size_t)n * K + k0 + 8 * hi + 16);
  v16h f;
#pragma unroll
  for (int i = 0; i < 8; ++i) { f[i] = b0[i]; f[8 + i] = b1[i]; }
  return f;
}

// B fragment from global (generic GEMM); zero-pad n >= N.
__device__ __forceinline__ v16h load_b_frag(const _Float16* __restrict__ W,
                                            int N, int K, int n0, int k0, int lane) {
  int hi = lane >> 4;
  int n  = n0 + (lane & 15);
  v16h f;
  if (n < N) {
    v8h b0 = *(const v8h*)(W + (size_t)n * K + k0 + 8 * hi);
    v8h b1 = *(const v8h*)(W + (size_t)n * K + k0 + 8 * hi + 16);
#pragma unroll
    for (int i = 0; i < 8; ++i) { f[i] = b0[i]; f[8 + i] = b1[i]; }
  } else {
#pragma unroll
    for (int i = 0; i < 16; ++i) f[i] = (_Float16)0.0f;
  }
  return f;
}

// A fragment (16-bit A 16x32 layout): halves e -> K = k0 + e + (e&8) + 8*hi.
__device__ __forceinline__ v16h load_a_frag(const _Float16* __restrict__ A,
                                            int K, int m, int k0, int hi) {
  v8h a0 = *(const v8h*)(A + (size_t)m * K + k0 + 8 * hi);
  v8h a1 = *(const v8h*)(A + (size_t)m * K + k0 + 8 * hi + 16);
  v16h f;
#pragma unroll
  for (int i = 0; i < 8; ++i) { f[i] = a0[i]; f[8 + i] = a1[i]; }
  return f;
}

// ---------------------------------------------------------------------------
// Generic GEMM: out = act(A(M,K) * W(N,K)^T + bias); 32x32 tile per wave.
// ---------------------------------------------------------------------------
__global__ void gemm_f16_kernel(const _Float16* __restrict__ A,
                                const _Float16* __restrict__ W,
                                const float* __restrict__ bias,
                                float* __restrict__ outF,
                                _Float16* __restrict__ outH,
                                int M, int N, int K, int relu) {
  int lane   = threadIdx.x & 31;
  int wave   = blockIdx.x * (blockDim.x >> 5) + (threadIdx.x >> 5);
  int tilesN = (N + 31) >> 5;
  int tilesM = (M + 31) >> 5;
  if (wave >= tilesM * tilesN) return;
  int tm = wave / tilesN, tn = wave % tilesN;
  int m0 = tm << 5, n0 = tn << 5;
  int hi = lane >> 4, lo = lane & 15;

  v8f acc00 = {}, acc01 = {}, acc10 = {}, acc11 = {};

  int mA0 = m0 + lo;       if (mA0 >= M) mA0 = M - 1;
  int mA1 = m0 + 16 + lo;  if (mA1 >= M) mA1 = M - 1;

  for (int k0 = 0; k0 < K; k0 += 32) {
    if (k0 + 32 < K) {   // prefetch next A k-slab (global_prefetch_b8)
      __builtin_prefetch(A + (size_t)mA0 * K + k0 + 32, 0, 1);
      __builtin_prefetch(A + (size_t)mA1 * K + k0 + 32, 0, 1);
    }
    v16h a0 = load_a_frag(A, K, mA0, k0, hi);
    v16h a1 = load_a_frag(A, K, mA1, k0, hi);
    v16h b0 = load_b_frag(W, N, K, n0, k0, lane);
    v16h b1 = load_b_frag(W, N, K, n0 + 16, k0, lane);
    acc00 = wmma16(a0, b0, acc00);
    acc01 = wmma16(a0, b1, acc01);
    acc10 = wmma16(a1, b0, acc10);
    acc11 = wmma16(a1, b1, acc11);
  }

#pragma unroll
  for (int sm = 0; sm < 2; ++sm) {
#pragma unroll
    for (int sn = 0; sn < 2; ++sn) {
      const v8f& acc = sm == 0 ? (sn == 0 ? acc00 : acc01)
                               : (sn == 0 ? acc10 : acc11);
      int n = n0 + sn * 16 + lo;
      if (n >= N) continue;
      float bv = bias ? bias[n] : 0.0f;
#pragma unroll
      for (int r = 0; r < 8; ++r) {
        int m = m0 + sm * 16 + r + 8 * hi;
        if (m >= M) continue;
        float v = acc[r] + bv;
        if (relu) v = fmaxf(v, 0.0f);
        if (outF) outF[(size_t)m * N + n] = v;
        if (outH) outH[(size_t)m * N + n] = (_Float16)v;
      }
    }
  }
}

// ---------------------------------------------------------------------------
// Conv output store: NCHW f16, fused bias + ReLU.
// ---------------------------------------------------------------------------
__device__ __forceinline__ void conv_store(const v8f& acc, _Float16* __restrict__ out,
                                           const float* __restrict__ bias,
                                           int m0, int n0, int lo, int hi,
                                           int Cout, int P) {
  int n = n0 + lo;
  float bv = bias[n];
#pragma unroll
  for (int r = 0; r < 8; ++r) {
    int m = m0 + r + 8 * hi;
    int f = m / P, p = m % P;
    float v = fmaxf(acc[r] + bv, 0.0f);
    out[((size_t)f * Cout + n) * P + p] = (_Float16)v;
  }
}

// ---------------------------------------------------------------------------
// conv1: x f32 (4,84,84) -> (32,20,20), 8x8 stride 4. K=256, k=c*64+r*8+s.
// An 8-aligned k-chunk is one (c,r) row of 8 contiguous pixels: 2x float4.
// ---------------------------------------------------------------------------
__device__ __forceinline__ v16h conv1_frag(const float* __restrict__ base,
                                           int k0, int hi) {
  int kb0 = k0 + (hi << 3);
  int kb1 = kb0 + 16;
  const float* r0 = base + (kb0 >> 6) * 7056 + ((kb0 >> 3) & 7) * 84;
  const float* r1 = base + (kb1 >> 6) * 7056 + ((kb1 >> 3) & 7) * 84;
  float4 A = *(const float4*)r0, B = *(const float4*)(r0 + 4);
  float4 C = *(const float4*)r1, D = *(const float4*)(r1 + 4);
  const float sc = 1.0f / 255.0f;
  v16h f;
  f[0]  = (_Float16)(A.x * sc); f[1]  = (_Float16)(A.y * sc);
  f[2]  = (_Float16)(A.z * sc); f[3]  = (_Float16)(A.w * sc);
  f[4]  = (_Float16)(B.x * sc); f[5]  = (_Float16)(B.y * sc);
  f[6]  = (_Float16)(B.z * sc); f[7]  = (_Float16)(B.w * sc);
  f[8]  = (_Float16)(C.x * sc); f[9]  = (_Float16)(C.y * sc);
  f[10] = (_Float16)(C.z * sc); f[11] = (_Float16)(C.w * sc);
  f[12] = (_Float16)(D.x * sc); f[13] = (_Float16)(D.y * sc);
  f[14] = (_Float16)(D.z * sc); f[15] = (_Float16)(D.w * sc);
  return f;
}

__global__ void conv1_kernel(const float* __restrict__ x,
                             const _Float16* __restrict__ Wg,
                             const float* __restrict__ bias,
                             _Float16* __restrict__ out, int M) {
  extern __shared__ _Float16 shW[];
  const int K = 256, Cout = 32, P = 400;
  stage_weights_lds(shW, Wg, Cout * K);   // async-to-LDS path
  __syncthreads();

  int lane = threadIdx.x & 31;
  int wave = blockIdx.x * (blockDim.x >> 5) + (threadIdx.x >> 5);
  if (wave >= (M >> 5)) return;
  int m0 = wave << 5;
  int hi = lane >> 4, lo = lane & 15;

  int mA0 = m0 + lo, mA1 = m0 + 16 + lo;
  int f0 = mA0 / 400, p0 = mA0 % 400;
  int f1 = mA1 / 400, p1 = mA1 % 400;
  const float* base0 = x + (size_t)f0 * 28224 + (p0 / 20) * 336 + (p0 % 20) * 4;
  const float* base1 = x + (size_t)f1 * 28224 + (p1 / 20) * 336 + (p1 % 20) * 4;

  v8f acc00 = {}, acc01 = {}, acc10 = {}, acc11 = {};
#pragma unroll
  for (int k0 = 0; k0 < K; k0 += 32) {
    v16h a0 = conv1_frag(base0, k0, hi);
    v16h a1 = conv1_frag(base1, k0, hi);
    v16h b0 = lds_b_frag(shW, K, 0, k0, lane);
    v16h b1 = lds_b_frag(shW, K, 16, k0, lane);
    acc00 = wmma16(a0, b0, acc00);
    acc01 = wmma16(a0, b1, acc01);
    acc10 = wmma16(a1, b0, acc10);
    acc11 = wmma16(a1, b1, acc11);
  }
  conv_store(acc00, out, bias, m0,      0,  lo, hi, Cout, P);
  conv_store(acc01, out, bias, m0,      16, lo, hi, Cout, P);
  conv_store(acc10, out, bias, m0 + 16, 0,  lo, hi, Cout, P);
  conv_store(acc11, out, bias, m0 + 16, 16, lo, hi, Cout, P);
}

// ---------------------------------------------------------------------------
// conv2: act1 f16 (32,20,20) -> (64,9,9), 4x4 stride 2. K=512, k=c*16+r*4+s.
// Weights staged by TDM. An 8-chunk = rows (rb, rb+1), each 4 contiguous
// halves: 4x dword loads.
// ---------------------------------------------------------------------------
__device__ __forceinline__ void ld4h(const _Float16* __restrict__ p, v16h& f, int e) {
  const unsigned* up = (const unsigned*)p;
  union { unsigned u; _Float16 h[2]; } c0, c1;
  c0.u = up[0]; c1.u = up[1];
  f[e] = c0.h[0]; f[e + 1] = c0.h[1]; f[e + 2] = c1.h[0]; f[e + 3] = c1.h[1];
}

__global__ void conv2_kernel(const _Float16* __restrict__ act1,
                             const _Float16* __restrict__ Wg,
                             const float* __restrict__ bias,
                             _Float16* __restrict__ out, int M) {
  extern __shared__ _Float16 shW[];
  const int K = 512, Cout = 64, P = 81;
  if ((threadIdx.x >> 5) == 0)
    tdm_stage_weights(Wg, (unsigned)K, (unsigned)Cout, (unsigned)K);
  __syncthreads();

  int lane = threadIdx.x & 31;
  int wave = blockIdx.x * (blockDim.x >> 5) + (threadIdx.x >> 5);
  if (wave >= (M >> 5) * 2) return;
  int tm = wave >> 1, tn = wave & 1;
  int m0 = tm << 5, n0 = tn << 5;
  int hi = lane >> 4, lo = lane & 15;

  int mA0 = m0 + lo, mA1 = m0 + 16 + lo;
  int f0 = mA0 / 81, p0 = mA0 % 81;
  int f1 = mA1 / 81, p1 = mA1 % 81;
  const _Float16* base0 = act1 + (size_t)f0 * 12800 + (p0 / 9) * 40 + (p0 % 9) * 2;
  const _Float16* base1 = act1 + (size_t)f1 * 12800 + (p1 / 9) * 40 + (p1 % 9) * 2;

  v8f acc00 = {}, acc01 = {}, acc10 = {}, acc11 = {};
  for (int k0 = 0; k0 < K; k0 += 32) {
    v16h a0, a1;
#pragma unroll
    for (int ch = 0; ch < 2; ++ch) {
      int kb = k0 + (hi << 3) + ch * 16;
      int c = kb >> 4, rb = (kb >> 2) & 3;
      const _Float16* r0 = base0 + c * 400 + rb * 20;
      const _Float16* r1 = base1 + c * 400 + rb * 20;
      ld4h(r0, a0, ch * 8);  ld4h(r0 + 20, a0, ch * 8 + 4);
      ld4h(r1, a1, ch * 8);  ld4h(r1 + 20, a1, ch * 8 + 4);
    }
    v16h b0 = lds_b_frag(shW, K, n0, k0, lane);
    v16h b1 = lds_b_frag(shW, K, n0 + 16, k0, lane);
    acc00 = wmma16(a0, b0, acc00);
    acc01 = wmma16(a0, b1, acc01);
    acc10 = wmma16(a1, b0, acc10);
    acc11 = wmma16(a1, b1, acc11);
  }
  conv_store(acc00, out, bias, m0,      n0,      lo, hi, Cout, P);
  conv_store(acc01, out, bias, m0,      n0 + 16, lo, hi, Cout, P);
  conv_store(acc10, out, bias, m0 + 16, n0,      lo, hi, Cout, P);
  conv_store(acc11, out, bias, m0 + 16, n0 + 16, lo, hi, Cout, P);
}

// ---------------------------------------------------------------------------
// conv3: act2 f16 (64,9,9) -> (64,7,7), 3x3 stride 1. K=576, k=c*9+r*3+s.
// Weights staged by TDM. 9 not 8-aligned -> scalar gather (hoisted base).
// ---------------------------------------------------------------------------
__global__ void conv3_kernel(const _Float16* __restrict__ act2,
                             const _Float16* __restrict__ Wg,
                             const float* __restrict__ bias,
                             _Float16* __restrict__ out, int M) {
  extern __shared__ _Float16 shW[];
  const int K = 576, Cout = 64, P = 49;
  if ((threadIdx.x >> 5) == 0)
    tdm_stage_weights(Wg, (unsigned)K, (unsigned)Cout, (unsigned)K);
  __syncthreads();

  int lane = threadIdx.x & 31;
  int wave = blockIdx.x * (blockDim.x >> 5) + (threadIdx.x >> 5);
  if (wave >= (M >> 5) * 2) return;
  int tm = wave >> 1, tn = wave & 1;
  int m0 = tm << 5, n0 = tn << 5;
  int hi = lane >> 4, lo = lane & 15;

  int mA0 = m0 + lo, mA1 = m0 + 16 + lo;
  int f0 = mA0 / 49, p0 = mA0 % 49;
  int f1 = mA1 / 49, p1 = mA1 % 49;
  const _Float16* base0 = act2 + (size_t)f0 * 5184 + (p0 / 7) * 9 + (p0 % 7);
  const _Float16* base1 = act2 + (size_t)f1 * 5184 + (p1 / 7) * 9 + (p1 % 7);

  v8f acc00 = {}, acc01 = {}, acc10 = {}, acc11 = {};
  for (int k0 = 0; k0 < K; k0 += 32) {
    v16h a0, a1;
#pragma unroll
    for (int e = 0; e < 16; ++e) {
      int k = k0 + e + (e & 8) + (hi << 3);
      int c = k / 9, q = k % 9;
      int off = c * 81 + (q / 3) * 9 + (q % 3);
      a0[e] = base0[off];
      a1[e] = base1[off];
    }
    v16h b0 = lds_b_frag(shW, K, n0, k0, lane);
    v16h b1 = lds_b_frag(shW, K, n0 + 16, k0, lane);
    acc00 = wmma16(a0, b0, acc00);
    acc01 = wmma16(a0, b1, acc01);
    acc10 = wmma16(a1, b0, acc10);
    acc11 = wmma16(a1, b1, acc11);
  }
  conv_store(acc00, out, bias, m0,      n0,      lo, hi, Cout, P);
  conv_store(acc01, out, bias, m0,      n0 + 16, lo, hi, Cout, P);
  conv_store(acc10, out, bias, m0 + 16, n0,      lo, hi, Cout, P);
  conv_store(acc11, out, bias, m0 + 16, n0 + 16, lo, hi, Cout, P);
}

// ---------------------------------------------------------------------------
// f32 -> f16 weight conversion
// ---------------------------------------------------------------------------
__global__ void cvt_f16_kernel(const float* __restrict__ in,
                               _Float16* __restrict__ out, int n) {
  int i = blockIdx.x * blockDim.x + threadIdx.x;
  if (i < n) out[i] = (_Float16)in[i];
}

// ---------------------------------------------------------------------------
// Depthwise causal conv1d (kernel 4) + bias + SiLU.
// ---------------------------------------------------------------------------
__global__ void dwconv_silu_kernel(const float* __restrict__ xz,
                                   const float* __restrict__ w,
                                   const float* __restrict__ bias,
                                   float* __restrict__ uF,
                                   _Float16* __restrict__ uH) {
  int idx = blockIdx.x * blockDim.x + threadIdx.x;
  if (idx >= NFRAMES * D_INNER) return;
  int row = idx >> 8;
  int d   = idx & 255;
  int b = row >> 5, s = row & 31;
  float acc = bias[d];
#pragma unroll
  for (int j = 0; j < 4; ++j) {
    int ss = s - 3 + j;
    if (ss >= 0) acc += w[d * 4 + j] * xz[(size_t)((b << 5) + ss) * 512 + d];
  }
  float v = acc * sigmoidf_(acc);
  uF[idx] = v;
  uH[idx] = (_Float16)v;
}

// ---------------------------------------------------------------------------
// Selective scan: one thread per (b, d); 32-wide state in registers.
// ---------------------------------------------------------------------------
__global__ void scan_kernel(const float* __restrict__ xdbl,
                            const float* __restrict__ uF,
                            const float* __restrict__ xz,
                            const float* __restrict__ dtw,
                            const float* __restrict__ dtb,
                            const float* __restrict__ A_log,
                            const float* __restrict__ Dvec,
                            _Float16* __restrict__ yLast) {
  int idx = blockIdx.x * blockDim.x + threadIdx.x;
  if (idx >= 64 * D_INNER) return;
  int b = idx >> 8;
  int d = idx & 255;

  float Arow[D_STATE], h[D_STATE];
#pragma unroll
  for (int n = 0; n < D_STATE; ++n) {
    Arow[n] = -__expf(A_log[d * D_STATE + n]);
    h[n] = 0.0f;
  }
  float dw[DT_RANK];
#pragma unroll
  for (int r = 0; r < DT_RANK; ++r) dw[r] = dtw[d * DT_RANK + r];
  float dtbv = dtb[d];
  float Dd   = Dvec[d];
  float ylast = 0.0f;

  for (int s = 0; s < 32; ++s) {
    int row = (b << 5) + s;
    const float* xd = xdbl + (size_t)row * 72;
    float dt = dtbv;
#pragma unroll
    for (int r = 0; r < DT_RANK; ++r) dt += xd[r] * dw[r];
    dt = (dt > 20.0f) ? dt : log1pf(__expf(dt));
    float u = uF[(size_t)row * D_INNER + d];
    float y = 0.0f;
#pragma unroll
    for (int n = 0; n < D_STATE; ++n) {
      float dA  = __expf(dt * Arow[n]);
      float dBu = dt * xd[8 + n] * u;
      h[n] = dA * h[n] + dBu;
      y += h[n] * xd[40 + n];
    }
    y += Dd * u;
    float z = xz[(size_t)row * 512 + 256 + d];
    y *= z * sigmoidf_(z);
    if (s == 31) ylast = y;
  }
  yLast[(size_t)b * D_INNER + d] = (_Float16)ylast;
}

// ---------------------------------------------------------------------------
// Host launcher
// ---------------------------------------------------------------------------
extern "C" void kernel_launch(void* const* d_in, const int* in_sizes, int n_in,
                              void* d_out, int out_size, void* d_ws, size_t ws_size,
                              hipStream_t stream) {
  const float* x         = (const float*)d_in[0];
  const float* conv1_w   = (const float*)d_in[1];
  const float* conv1_b   = (const float*)d_in[2];
  const float* conv2_w   = (const float*)d_in[3];
  const float* conv2_b   = (const float*)d_in[4];
  const float* conv3_w   = (const float*)d_in[5];
  const float* conv3_b   = (const float*)d_in[6];
  const float* feat_w    = (const float*)d_in[7];
  const float* feat_b    = (const float*)d_in[8];
  const float* in_proj_w = (const float*)d_in[9];
  const float* conv1d_w  = (const float*)d_in[10];
  const float* conv1d_b  = (const float*)d_in[11];
  const float* x_proj_w  = (const float*)d_in[12];
  const float* dt_proj_w = (const float*)d_in[13];
  const float* dt_proj_b = (const float*)d_in[14];
  const float* A_log     = (const float*)d_in[15];
  const float* Dvec      = (const float*)d_in[16];
  const float* out_proj_w= (const float*)d_in[17];
  const float* fc1_w     = (const float*)d_in[18];
  const float* fc1_b     = (const float*)d_in[19];
  const float* fc2_w     = (const float*)d_in[20];
  const float* fc2_b     = (const float*)d_in[21];
  const float* q_w       = (const float*)d_in[22];
  const float* q_b       = (const float*)d_in[23];

  float* outQ      = (float*)d_out;            // (64,18)
  float* outLatent = (float*)d_out + 64 * 18;  // (64,128)

  char* base = (char*)d_ws;
  size_t off = 0;
  auto alloc = [&](size_t bytes) -> void* {
    void* p = base + off;
    off = (off + bytes + 255) & ~(size_t)255;
    return p;
  };
  _Float16* W1h   = (_Float16*)alloc(8192 * 2);
  _Float16* W2h   = (_Float16*)alloc(32768 * 2);
  _Float16* W3h   = (_Float16*)alloc(36864 * 2);
  _Float16* WfH   = (_Float16*)alloc(401408 * 2);
  _Float16* WiH   = (_Float16*)alloc(131072 * 2);
  _Float16* WxH   = (_Float16*)alloc(18432 * 2);
  _Float16* WoH   = (_Float16*)alloc(32768 * 2);
  _Float16* Wf1H  = (_Float16*)alloc(16384 * 2);
  _Float16* Wf2H  = (_Float16*)alloc(16384 * 2);
  _Float16* WqH   = (_Float16*)alloc(2304 * 2);
  _Float16* act1  = (_Float16*)alloc((size_t)NFRAMES * 32 * 400 * 2);
  _Float16* act2  = (_Float16*)alloc((size_t)NFRAMES * 64 * 81 * 2);
  _Float16* act3  = (_Float16*)alloc((size_t)NFRAMES * 64 * 49 * 2);
  _Float16* featH = (_Float16*)alloc((size_t)NFRAMES * D_MODEL * 2);
  float*    xzF   = (float*)alloc((size_t)NFRAMES * 512 * 4);
  float*    uF    = (float*)alloc((size_t)NFRAMES * D_INNER * 4);
  _Float16* uH    = (_Float16*)alloc((size_t)NFRAMES * D_INNER * 2);
  float*    xdbl  = (float*)alloc((size_t)NFRAMES * 72 * 4);
  _Float16* yLast = (_Float16*)alloc((size_t)64 * D_INNER * 2);
  _Float16* latH  = (_Float16*)alloc((size_t)64 * D_MODEL * 2);
  _Float16* h1H   = (_Float16*)alloc((size_t)64 * 128 * 2);
  _Float16* h2H   = (_Float16*)alloc((size_t)64 * 128 * 2);
  (void)ws_size; (void)in_sizes; (void)n_in; (void)out_size;

  auto cvt = [&](const float* src, _Float16* dst, int n) {
    cvt_f16_kernel<<<(n + 255) / 256, 256, 0, stream>>>(src, dst, n);
  };
  cvt(conv1_w, W1h, 8192);
  cvt(conv2_w, W2h, 32768);
  cvt(conv3_w, W3h, 36864);
  cvt(feat_w, WfH, 401408);
  cvt(in_proj_w, WiH, 131072);
  cvt(x_proj_w, WxH, 18432);
  cvt(out_proj_w, WoH, 32768);
  cvt(fc1_w, Wf1H, 16384);
  cvt(fc2_w, Wf2H, 16384);
  cvt(q_w, WqH, 2304);

  auto gemm_blocks = [](int M, int N) {
    int waves = ((M + 31) / 32) * ((N + 31) / 32);
    return (waves + 7) / 8;
  };

  // conv1: M = 2048*400, Cout=32, K=256; W tile 16KB in LDS (async path)
  {
    int M = NFRAMES * 400;
    conv1_kernel<<<gemm_blocks(M, 32), 256, 32 * 256 * 2, stream>>>(
        x, W1h, conv1_b, act1, M);
  }
  // conv2: M = 2048*81, Cout=64, K=512; W tile 64KB in LDS (TDM path)
  {
    int M = NFRAMES * 81;
    conv2_kernel<<<gemm_blocks(M, 64), 256, 64 * 512 * 2, stream>>>(
        act1, W2h, conv2_b, act2, M);
  }
  // conv3: M = 2048*49, Cout=64, K=576; W tile 72KB in LDS (TDM path)
  {
    int M = NFRAMES * 49;
    conv3_kernel<<<gemm_blocks(M, 64), 256, 64 * 576 * 2, stream>>>(
        act2, W3h, conv3_b, act3, M);
  }
  // feat: (2048,3136) @ (3136,128) + bias -> f16
  gemm_f16_kernel<<<gemm_blocks(NFRAMES, 128), 256, 0, stream>>>(
      act3, WfH, feat_b, nullptr, featH, NFRAMES, 128, 3136, 0);
  // in_proj: (2048,128) @ (128,512) -> f32 xz
  gemm_f16_kernel<<<gemm_blocks(NFRAMES, 512), 256, 0, stream>>>(
      featH, WiH, nullptr, xzF, nullptr, NFRAMES, 512, 128, 0);
  // depthwise causal conv1d + SiLU
  dwconv_silu_kernel<<<(NFRAMES * D_INNER + 255) / 256, 256, 0, stream>>>(
      xzF, conv1d_w, conv1d_b, uF, uH);
  // x_proj: (2048,256) @ (256,72) -> f32 x_dbl
  gemm_f16_kernel<<<gemm_blocks(NFRAMES, 72), 256, 0, stream>>>(
      uH, WxH, nullptr, xdbl, nullptr, NFRAMES, 72, 256, 0);
  // selective scan (keeps only last timestep's gated y)
  scan_kernel<<<(64 * D_INNER + 255) / 256, 256, 0, stream>>>(
      xdbl, uF, xzF, dt_proj_w, dt_proj_b, A_log, Dvec, yLast);
  // out_proj on last timestep: (64,256) @ (256,128) -> latent + f16
  gemm_f16_kernel<<<gemm_blocks(64, 128), 256, 0, stream>>>(
      yLast, WoH, nullptr, outLatent, latH, 64, 128, 256, 0);
  // fc1 + ReLU
  gemm_f16_kernel<<<gemm_blocks(64, 128), 256, 0, stream>>>(
      latH, Wf1H, fc1_b, nullptr, h1H, 64, 128, 128, 1);
  // fc2 + ReLU
  gemm_f16_kernel<<<gemm_blocks(64, 128), 256, 0, stream>>>(
      h1H, Wf2H, fc2_b, nullptr, h2H, 64, 128, 128, 1);
  // q head: (64,128) @ (128,18) + bias -> f32 to d_out
  gemm_f16_kernel<<<gemm_blocks(64, 18), 256, 0, stream>>>(
      h2H, WqH, q_b, outQ, nullptr, 64, 18, 128, 0);
}
